// GCNEncoder_13271448945348
// MI455X (gfx1250) — matrix operand order repaired and checked by reading, not compiled, not run on hardware
//
#include <hip/hip_runtime.h>

// ---------------------------------------------------------------------------
// GCN (PyG semantics): per layer  h = relu( D^-1/2 (A+I) D^-1/2 (h @ W) + b )
//  - dense transform: v_wmma_f32_16x16x32_bf16, tiles fed by the Tensor Data
//    Mover (tensor_load_to_lds, TENSORcnt) -- no per-element VALU in the loop
//  - message passing: coalesced gathers + global f32 atomic adds (HBM bound)
// ---------------------------------------------------------------------------

typedef __attribute__((ext_vector_type(16))) __bf16 v16bf;
typedef __attribute__((ext_vector_type(8)))  float  v8f;
typedef __attribute__((ext_vector_type(4)))  unsigned int v4u;
typedef __attribute__((ext_vector_type(8)))  int v8i;
typedef __attribute__((ext_vector_type(4)))  int v4i;

union FragAB { v16bf v; uint4 q[2]; };
union FragC  { v8f   v; float f[8]; };

__device__ __forceinline__ unsigned short f2bf(float x) {
  unsigned int u = __float_as_uint(x);
  u += 0x7FFFu + ((u >> 16) & 1u);  // round-to-nearest-even
  return (unsigned short)(u >> 16);
}

// ---------------------------------------------------------------------------
// TDM: 2D bf16 tile (tile_d1 rows x tile_d0 elems) from global -> LDS.
// D# layout per CDNA5 ISA ch.8 (group0: count/lds/global/type, group1:
// data_size + dims/strides). Rows past tensor_d1 are OOB -> zero-filled.
// This toolchain exposes the 6-arg builtin: (g0, g1, g2, g3, g4, cpol).
// ---------------------------------------------------------------------------
__device__ __forceinline__ void tdm_load_2d(
    unsigned lds_addr, const unsigned short* gptr,
    unsigned tensor_d0, unsigned tensor_d1,
    unsigned tile_d0, unsigned tile_d1, unsigned stride0) {
  unsigned long long ga = (unsigned long long)(uintptr_t)gptr;
  v4u g0;
  g0[0] = 1u;                                   // count=1 (valid user D#)
  g0[1] = lds_addr;                             // lds_addr      [63:32]
  g0[2] = (unsigned)(ga & 0xFFFFFFFFu);         // global_addr   [95:64]
  g0[3] = (unsigned)((ga >> 32) & 0x01FFFFFFu)  // global_addr   [120:96]
          | (2u << 30);                         // type=2 image  [127:126]
  v8i g1;
  g1[0] = (int)(1u << 16);                      // data_size=1 -> 2 bytes
  g1[1] = (int)((tensor_d0 & 0xFFFFu) << 16);   // tensor_dim0[15:0] @ [63:48]
  g1[2] = (int)((tensor_d0 >> 16) | ((tensor_d1 & 0xFFFFu) << 16));
  g1[3] = (int)((tensor_d1 >> 16) | (tile_d0 << 16));   // tile_dim0 [127:112]
  g1[4] = (int)(tile_d1 & 0xFFFFu);             // tile_dim1 (tile_dim2=0)
  g1[5] = (int)stride0;                         // tensor_dim0_stride[31:0]
  g1[6] = 0;                                    // stride hi / dim1_stride lo
  g1[7] = 0;
  v4i z4 = {0, 0, 0, 0};
  v8i z8 = {0, 0, 0, 0, 0, 0, 0, 0};
  __builtin_amdgcn_tensor_load_to_lds(g0, g1, z4, z4, z8, 0);
}

// ---------------------------------------------------------------------------
// GEMM: C[M,256] = A[M,256](bf16) * B[256,256], Bt supplied bf16 TRANSPOSED
// ([col][k]).  Block tile 128x128, K-chunks of 32.  256 threads = 8 waves in
// a 4(M)x2(N) grid; each wave owns 32x64 = 2x4 WMMA accumulators.
// ---------------------------------------------------------------------------
constexpr int TM = 128, TN = 128, TK = 32;

__global__ __launch_bounds__(256) void gemm_bf16_wmma_tdm(
    const unsigned short* __restrict__ A,   // bf16 [M,256] row-major
    const unsigned short* __restrict__ Bt,  // bf16 [256,256] col-major (N,K)
    float* __restrict__ C, int M) {
  __shared__ __align__(16) unsigned short As[TM * TK];  // [row][k]
  __shared__ __align__(16) unsigned short Bs[TN * TK];  // [col][k]

  const int m0   = blockIdx.x * TM;
  const int n0   = blockIdx.y * TN;
  const int t    = threadIdx.x;
  const int lane = t & 31;
  const int wave = t >> 5;
  const int wm   = wave >> 1;   // 0..3 -> rows [wm*32, +32)
  const int wn   = wave & 1;    // 0..1 -> cols [wn*64, +64)
  const int lrow = lane & 15;
  const int hi   = lane >> 4;   // lane half selects K sub-range (ISA layout)
  const unsigned rows_left = (unsigned)(M - m0);

  FragC acc[2][4];
#pragma unroll
  for (int i = 0; i < 2; ++i)
#pragma unroll
    for (int j = 0; j < 4; ++j)
#pragma unroll
      for (int r = 0; r < 8; ++r) acc[i][j].f[r] = 0.0f;

  for (int k0 = 0; k0 < 256; k0 += TK) {
    if (wave == 0) {
      tdm_load_2d((unsigned)(uintptr_t)As, A + (size_t)m0 * 256 + k0,
                  256u, rows_left, TK, TM, 256u);
      tdm_load_2d((unsigned)(uintptr_t)Bs, Bt + (size_t)n0 * 256 + k0,
                  256u, 256u, TK, TN, 256u);
      __builtin_amdgcn_s_wait_tensorcnt(0);
    }
    __syncthreads();

    // A frag (16-bit 16x32 layout): lanes 0-15 hold K 0-7 & 16-23,
    // lanes 16-31 hold K 8-15 & 24-31, M = lane&15.
    FragAB a[2], b[4];
#pragma unroll
    for (int i = 0; i < 2; ++i) {
      const uint4* p = (const uint4*)&As[(wm * 32 + i * 16 + lrow) * TK];
      a[i].q[0] = p[hi];
      a[i].q[1] = p[2 + hi];
    }
    // B frag (16-bit 32x16 layout): lanes 0-15 hold K 0-7 & 16-23, lanes 16-31
    // hold K 8-15 & 24-31, N = lane&15 (same K order as A).
#pragma unroll
    for (int j = 0; j < 4; ++j) {
      const uint4* p = (const uint4*)&Bs[(wn * 64 + j * 16 + lrow) * TK];
      b[j].q[0] = p[hi];
      b[j].q[1] = p[2 + hi];
    }
#pragma unroll
    for (int i = 0; i < 2; ++i)
#pragma unroll
      for (int j = 0; j < 4; ++j)
        acc[i][j].v = __builtin_amdgcn_wmma_f32_16x16x32_bf16(
            false, a[i].v, false, b[j].v, (short)0, acc[i][j].v, false, false);
    __syncthreads();
  }

  // f32 C/D layout: VGPR r -> M = r (lanes 0-15) / r+8 (lanes 16-31), N=lane&15
#pragma unroll
  for (int i = 0; i < 2; ++i)
#pragma unroll
    for (int j = 0; j < 4; ++j) {
      const int coln = n0 + wn * 64 + j * 16 + lrow;
#pragma unroll
      for (int r = 0; r < 8; ++r) {
        const int rowm = m0 + wm * 32 + i * 16 + hi * 8 + r;
        if (rowm < M) C[(size_t)rowm * 256 + coln] = acc[i][j].f[r];
      }
    }
}

// ---------------------------------------------------------------------------
// Precision-conversion kernels (tiny, bandwidth-trivial)
// ---------------------------------------------------------------------------
__global__ void conv_f32_bf16(const float* __restrict__ in,
                              unsigned short* __restrict__ out, int n) {
  int i = blockIdx.x * blockDim.x + threadIdx.x;
  if (i < n) out[i] = f2bf(in[i]);
}
// W[l][k][c] -> Wt[l][c][k]  (bf16, transposed once; 3*256*256 elements)
__global__ void conv_W_transpose(const float* __restrict__ W,
                                 unsigned short* __restrict__ Wt, int total) {
  int idx = blockIdx.x * blockDim.x + threadIdx.x;
  if (idx >= total) return;
  int l = idx >> 16;
  int r = idx & 0xFFFF;
  int k = r >> 8;
  int c = r & 255;
  Wt[(size_t)l * 65536 + c * 256 + k] = f2bf(W[idx]);
}

// ---------------------------------------------------------------------------
// Degree / normalization
// ---------------------------------------------------------------------------
__global__ void init_deg(float* deg, int N) {
  int i = blockIdx.x * blockDim.x + threadIdx.x;
  if (i < N) deg[i] = 1.0f;  // self-loop
}
__global__ void accum_deg(const int* __restrict__ dst, float* deg, int E) {
  int e = blockIdx.x * blockDim.x + threadIdx.x;
  if (e < E) atomicAdd(&deg[dst[e]], 1.0f);
}
__global__ void finalize_dinv(float* deg, int N) {
  int i = blockIdx.x * blockDim.x + threadIdx.x;
  if (i < N) {
    float d = deg[i];
    deg[i] = d > 0.0f ? rsqrtf(d) : 0.0f;
  }
}

// agg[i,:] = dinv[i]^2 * t[i,:]   (self-loop message)
__global__ __launch_bounds__(256) void agg_init(
    const float* __restrict__ t, const float* __restrict__ dinv,
    float* __restrict__ agg) {
  const int node = blockIdx.x;
  const int c    = threadIdx.x;
  const float di = dinv[node];
  agg[(size_t)node * 256 + c] = di * di * t[(size_t)node * 256 + c];
}

// One wave per edge; lanes stride the 256-wide feature row.
__global__ __launch_bounds__(256) void edge_scatter(
    const int* __restrict__ src, const int* __restrict__ dst,
    const float* __restrict__ dinv, const float* __restrict__ t,
    float* __restrict__ agg, int E) {
  const int gid  = blockIdx.x * blockDim.x + threadIdx.x;
  const int e    = gid >> 5;
  const int lane = gid & 31;
  if (e >= E) return;
  const int s = src[e];
  const int d = dst[e];
  const float w = dinv[s] * dinv[d];
  const float* trow = t + (size_t)s * 256;
  float* arow = agg + (size_t)d * 256;
#pragma unroll
  for (int c = lane; c < 256; c += 32) atomicAdd(&arow[c], w * trow[c]);
}

// Epilogue fused with next layer's input precision: relu(agg+b) -> bf16
__global__ __launch_bounds__(256) void bias_relu_bf16(
    const float* __restrict__ agg, const float* __restrict__ bias,
    unsigned short* __restrict__ h) {
  const int node = blockIdx.x;
  const int c    = threadIdx.x;
  const float v  = agg[(size_t)node * 256 + c] + bias[c];
  h[(size_t)node * 256 + c] = f2bf(v > 0.0f ? v : 0.0f);
}
// Final layer: relu(agg+b) -> f32 output
__global__ __launch_bounds__(256) void bias_relu_f32(
    const float* __restrict__ agg, const float* __restrict__ bias,
    float* __restrict__ h) {
  const int node = blockIdx.x;
  const int c    = threadIdx.x;
  const float v  = agg[(size_t)node * 256 + c] + bias[c];
  h[(size_t)node * 256 + c] = v > 0.0f ? v : 0.0f;
}

// ---------------------------------------------------------------------------
extern "C" void kernel_launch(void* const* d_in, const int* in_sizes, int n_in,
                              void* d_out, int out_size, void* d_ws, size_t ws_size,
                              hipStream_t stream) {
  const float* x  = (const float*)d_in[0];  // [N,256]
  const int*   ei = (const int*)d_in[1];    // [2,E]
  const float* W  = (const float*)d_in[2];  // [L,256,256]
  const float* b  = (const float*)d_in[3];  // [L,256]

  const int N = in_sizes[0] / 256;
  const int E = in_sizes[1] / 2;
  const int L = in_sizes[2] / (256 * 256);

  const int* src = ei;
  const int* dst = ei + E;

  // workspace: dinv[N] f32 | agg[N*256] f32 | hb[N*256] bf16 | Wt[L*64K] bf16
  float* dinv = (float*)d_ws;
  float* agg  = dinv + N;
  unsigned short* hb = (unsigned short*)(agg + (size_t)N * 256);
  unsigned short* Wt = hb + (size_t)N * 256;
  float* tbuf = (float*)d_out;  // GEMM output buffer; overwritten at the end

  // --- one-time precision prep --------------------------------------------
  conv_f32_bf16<<<(N * 256 + 255) / 256, 256, 0, stream>>>(x, hb, N * 256);
  conv_W_transpose<<<(L * 65536 + 255) / 256, 256, 0, stream>>>(W, Wt, L * 65536);

  // --- normalization: dinv = rsqrt(deg(A+I)) -------------------------------
  init_deg<<<(N + 255) / 256, 256, 0, stream>>>(dinv, N);
  accum_deg<<<(E + 255) / 256, 256, 0, stream>>>(dst, dinv, E);
  finalize_dinv<<<(N + 255) / 256, 256, 0, stream>>>(dinv, N);

  // --- layers --------------------------------------------------------------
  const dim3 ggrid((N + TM - 1) / TM, 256 / TN);
  for (int l = 0; l < L; ++l) {
    gemm_bf16_wmma_tdm<<<ggrid, 256, 0, stream>>>(
        hb, Wt + (size_t)l * 65536, tbuf, N);
    agg_init<<<N, 256, 0, stream>>>(tbuf, dinv, agg);
    edge_scatter<<<(E + 7) / 8, 256, 0, stream>>>(src, dst, dinv, tbuf, agg, E);
    if (l == L - 1) {
      bias_relu_f32<<<N, 256, 0, stream>>>(agg, b + (size_t)l * 256,
                                           (float*)d_out);
    } else {
      bias_relu_bf16<<<N, 256, 0, stream>>>(agg, b + (size_t)l * 256, hb);
    }
  }
}
